// Model_61091614818850
// MI455X (gfx1250) — 8-bit matrix operand order repaired and checked by reading, compiled, not run on hardware
//
#include <hip/hip_runtime.h>
#include <stdint.h>

typedef __attribute__((ext_vector_type(8))) int v8i;

#define EPSBN 1e-5f

__device__ __forceinline__ float wave_max32(float v) {
#pragma unroll
  for (int off = 16; off > 0; off >>= 1)
    v = fmaxf(v, __shfl_xor(v, off, 32));
  return v;
}

__global__ __launch_bounds__(64) void k_init_slots(float* slots) {
  slots[threadIdx.x] = 0.0f;
}

__global__ __launch_bounds__(64) void k_bnfold(const float* __restrict__ g,
                                               const float* __restrict__ b,
                                               const float* __restrict__ m,
                                               const float* __restrict__ v,
                                               float* __restrict__ inv,
                                               float* __restrict__ bias, int C) {
  int c = blockIdx.x * blockDim.x + threadIdx.x;
  if (c >= C) return;
  float iv = g[c] * rsqrtf(v[c] + EPSBN);
  inv[c] = iv;
  bias[c] = b[c] - m[c] * iv;
}

__global__ __launch_bounds__(64) void k_wq_dw(const float* __restrict__ w,
    const float* __restrict__ g, const float* __restrict__ b,
    const float* __restrict__ m, const float* __restrict__ v,
    signed char* __restrict__ code, float* __restrict__ cScale,
    float* __restrict__ cBias, int C, int oneBit) {
  int c = blockIdx.x * blockDim.x + threadIdx.x;
  if (c >= C) return;
  float asum = 0.0f, amax = 0.0f, wv[9];
#pragma unroll
  for (int k = 0; k < 9; ++k) {
    wv[k] = w[c * 9 + k];
    float a = fabsf(wv[k]);
    asum += a;
    amax = fmaxf(amax, a);
  }
  float s1 = fmaxf(asum * (1.0f / 9.0f), 1e-8f);
  float s2 = fmaxf(amax, 1e-8f);
  bool one = c < oneBit;
  float scale = one ? s1 : s2;
#pragma unroll
  for (int k = 0; k < 9; ++k) {
    int qv;
    if (one) qv = (wv[k] > 0.0f) ? 1 : ((wv[k] < 0.0f) ? -1 : 0);
    else {
      float r = rintf(wv[k] / s2);
      qv = (int)fminf(fmaxf(r, -1.0f), 1.0f);
    }
    code[c * 9 + k] = (signed char)qv;
  }
  float iv = g[c] * rsqrtf(v[c] + EPSBN);
  cScale[c] = scale * iv;
  cBias[c] = b[c] - m[c] * iv;
}

__global__ __launch_bounds__(64) void k_wq_pw(const float* __restrict__ w,
    const float* __restrict__ g, const float* __restrict__ b,
    const float* __restrict__ m, const float* __restrict__ v,
    signed char* __restrict__ code, float* __restrict__ cScale,
    float* __restrict__ cBias, int O, int K, int Kpad, int oneBit) {
  int o = blockIdx.x * blockDim.x + threadIdx.x;
  if (o >= O) return;
  float asum = 0.0f, amax = 0.0f;
  const float* wr = w + (size_t)o * K;
  for (int k = 0; k < K; ++k) {
    float a = fabsf(wr[k]);
    asum += a;
    amax = fmaxf(amax, a);
  }
  float s1 = fmaxf(asum / (float)K, 1e-8f);
  float s2 = fmaxf(amax, 1e-8f);
  bool one = o < oneBit;
  float scale = one ? s1 : s2;
  signed char* cr = code + (size_t)o * Kpad;
  for (int k = 0; k < Kpad; ++k) {
    int qv = 0;
    if (k < K) {
      float x = wr[k];
      if (one) qv = (x > 0.0f) ? 1 : ((x < 0.0f) ? -1 : 0);
      else {
        float r = rintf(x / s2);
        qv = (int)fminf(fmaxf(r, -1.0f), 1.0f);
      }
    }
    cr[k] = (signed char)qv;
  }
  float iv = g[o] * rsqrtf(v[o] + EPSBN);
  cScale[o] = scale * iv;
  cBias[o] = b[o] - m[o] * iv;
}

__global__ __launch_bounds__(256) void k_head(const float* __restrict__ x,
    const float* __restrict__ w, const float* __restrict__ inv,
    const float* __restrict__ bias, float* __restrict__ out,
    float* __restrict__ maxSlot, int total) {
  int idx = blockIdx.x * blockDim.x + threadIdx.x;
  float y = 0.0f;
  if (idx < total) {
    int c = idx & 31;
    int pix = idx >> 5;
    int ow = pix % 112;
    int t = pix / 112;
    int oh = t % 112;
    int n = t / 112;
    float acc = 0.0f;
#pragma unroll
    for (int kh = 0; kh < 3; ++kh) {
      int ih = oh * 2 - 1 + kh;
      if ((unsigned)ih < 224u) {
#pragma unroll
        for (int kw = 0; kw < 3; ++kw) {
          int iw = ow * 2 - 1 + kw;
          if ((unsigned)iw < 224u) {
#pragma unroll
            for (int ci = 0; ci < 3; ++ci)
              acc += x[((size_t)(n * 3 + ci) * 224 + ih) * 224 + iw] *
                     w[((c * 3 + ci) * 3 + kh) * 3 + kw];
          }
        }
      }
    }
    y = fmaxf(acc * inv[c] + bias[c], 0.0f);
    out[idx] = y;
  }
  y = wave_max32(y);
  if ((threadIdx.x & 31) == 0) atomicMax((int*)maxSlot, __float_as_int(y));
}

__global__ __launch_bounds__(256) void k_quant(const float* __restrict__ h,
    const float* __restrict__ maxIn, unsigned char* __restrict__ q,
    int C, int Kpad, long long total) {
  long long idx = (long long)blockIdx.x * blockDim.x + threadIdx.x;
  if (idx >= total) return;
  int c = (int)(idx % Kpad);
  long long pix = idx / Kpad;
  unsigned char val = 0;
  if (c < C) {
    float s = fmaxf(maxIn[0] * (1.0f / 255.0f), 1e-8f);
    float r = rintf(h[pix * C + c] / s);
    val = (unsigned char)fminf(r, 255.0f);
  }
  q[idx] = val;
}

__global__ __launch_bounds__(256) void k_dw(const unsigned char* __restrict__ actQ,
    const signed char* __restrict__ wQ, const float* __restrict__ cScale,
    const float* __restrict__ cBias, const float* __restrict__ maxIn,
    float* __restrict__ outF, float* __restrict__ maxOut,
    int C, int Kpad, int IH, int IW, int OH, int OW, int stride, long long total) {
  long long idx = (long long)blockIdx.x * blockDim.x + threadIdx.x;
  float y = 0.0f;
  if (idx < total) {
    int c = (int)(idx % C);
    long long pix = idx / C;
    int ow = (int)(pix % OW);
    long long t = pix / OW;
    int oh = (int)(t % OH);
    int n = (int)(t / OH);
    float sIn = fmaxf(maxIn[0] * (1.0f / 255.0f), 1e-8f);
    int acc = 0;
#pragma unroll
    for (int kh = 0; kh < 3; ++kh) {
      int ih = oh * stride - 1 + kh;
      if ((unsigned)ih < (unsigned)IH) {
#pragma unroll
        for (int kw = 0; kw < 3; ++kw) {
          int iw = ow * stride - 1 + kw;
          if ((unsigned)iw < (unsigned)IW)
            acc += (int)actQ[((size_t)(n * IH + ih) * IW + iw) * Kpad + c] *
                   (int)wQ[c * 9 + kh * 3 + kw];
        }
      }
    }
    y = fmaxf((float)acc * sIn * cScale[c] + cBias[c], 0.0f);
    outF[idx] = y;
  }
  y = wave_max32(y);
  if ((threadIdx.x & 31) == 0) atomicMax((int*)maxOut, __float_as_int(y));
}

__global__ __launch_bounds__(128) void k_pw_wmma(
    const unsigned char* __restrict__ actQ,
    const signed char* __restrict__ wQ,
    const float* __restrict__ cScale,
    const float* __restrict__ cBias,
    const float* __restrict__ maxIn,
    float* __restrict__ outF,
    float* __restrict__ maxOut,
    int M, int Kpad, int nTiles) {
  int nTile = blockIdx.y * blockDim.y + threadIdx.y;
  if (nTile >= nTiles) return;
  int mTile = blockIdx.x;
  int lane = threadIdx.x;
  int hi = lane >> 4;
  int l15 = lane & 15;

  float sIn = fmaxf(maxIn[0] * (1.0f / 255.0f), 1e-8f);

  const signed char* aRow = wQ + (size_t)(mTile * 16 + l15) * Kpad;
  const unsigned char* bCol = actQ + (size_t)(nTile * 16 + l15) * Kpad;

  v8i acc;
#pragma unroll
  for (int i = 0; i < 8; ++i) acc[i] = 0;

  for (int k0 = 0; k0 < Kpad; k0 += 64) {
    v8i a, b;
#pragma unroll
    for (int i = 0; i < 8; ++i)
      a[i] = *(const int*)(aRow + k0 + (i >> 1) * 16 + hi * 8 + (i & 1) * 4);
#pragma unroll
    for (int i = 0; i < 8; ++i)
      b[i] = *(const int*)(bCol + k0 + (i >> 1) * 16 + hi * 8 + (i & 1) * 4);
    acc = __builtin_amdgcn_wmma_i32_16x16x64_iu8(true, a, false, b, acc, false, false);
  }

  size_t p = (size_t)nTile * 16 + l15;
  float lmax = 0.0f;
#pragma unroll
  for (int r = 0; r < 8; ++r) {
    int m = mTile * 16 + 8 * hi + r;
    float y = fmaxf((float)acc[r] * sIn * cScale[m] + cBias[m], 0.0f);
    outF[p * (size_t)M + m] = y;
    lmax = fmaxf(lmax, y);
  }
  lmax = wave_max32(lmax);
  if (lane == 0) atomicMax((int*)maxOut, __float_as_int(lmax));
}

__global__ __launch_bounds__(256) void k_pool(const float* __restrict__ h,
                                              float* __restrict__ out) {
  int idx = blockIdx.x * blockDim.x + threadIdx.x;
  if (idx >= 32 * 1024) return;
  int c = idx & 1023;
  int n = idx >> 10;
  float s = 0.0f;
  for (int i = 0; i < 49; ++i) s += h[(((size_t)(n * 49 + i)) << 10) + c];
  out[idx] = s * (1.0f / 49.0f);
}

__global__ __launch_bounds__(256) void k_fc(const float* __restrict__ pooled,
    const float* __restrict__ w, const float* __restrict__ b,
    float* __restrict__ out) {
  int idx = blockIdx.x * blockDim.x + threadIdx.x;
  if (idx >= 32000) return;
  int o = idx % 1000;
  int n = idx / 1000;
  float acc = b[o];
  const float* pr = pooled + (size_t)n * 1024;
  const float* wr = w + (size_t)o * 1024;
  for (int k = 0; k < 1024; ++k) acc += pr[k] * wr[k];
  out[idx] = acc;
}

extern "C" void kernel_launch(void* const* d_in, const int* in_sizes, int n_in,
                              void* d_out, int out_size, void* d_ws, size_t ws_size,
                              hipStream_t stream) {
  static const int CIN[13]   = {32,64,128,128,256,256,512,512,512,512,512,512,1024};
  static const int COUT[13]  = {64,128,128,256,256,512,512,512,512,512,512,1024,1024};
  static const int STR[13]   = {1,2,1,2,1,2,1,1,1,1,1,2,1};
  static const int OB_IN[13] = {32,52,40,101,148,223,373,404,236,376,419,233,892};
  static const int OB_OUT[13]= {52,40,101,148,223,373,404,236,376,419,233,892,1024};
  const int NB = 32;
  if (n_in < 138) return;

  const float* x      = (const float*)d_in[0];
  const float* head_w = (const float*)d_in[1];

  uint8_t* ws = (uint8_t*)d_ws;
  size_t cur = 0;
  auto carve = [&](size_t bytes) -> uint8_t* {
    cur = (cur + 255) & ~(size_t)255;
    uint8_t* p = ws + cur;
    cur += bytes;
    return p;
  };

  float* slots    = (float*)carve(64 * sizeof(float));
  float* headInv  = (float*)carve(32 * sizeof(float));
  float* headBias = (float*)carve(32 * sizeof(float));

  signed char* dwCode[13]; float* dwScale[13]; float* dwBias[13];
  signed char* pwCode[13]; float* pwScale[13]; float* pwBias[13];
  int KPAD[13];
  for (int j = 0; j < 13; ++j) {
    int cin = CIN[j], cout = COUT[j];
    int kp = (cin < 64) ? 64 : cin;
    KPAD[j] = kp;
    dwCode[j]  = (signed char*)carve((size_t)cin * 9);
    dwScale[j] = (float*)carve((size_t)cin * 4);
    dwBias[j]  = (float*)carve((size_t)cin * 4);
    pwCode[j]  = (signed char*)carve((size_t)cout * kp);
    pwScale[j] = (float*)carve((size_t)cout * 4);
    pwBias[j]  = (float*)carve((size_t)cout * 4);
  }
  unsigned char* bufQ = (unsigned char*)carve((size_t)401408 * 64);
  float* bufB   = (float*)carve((size_t)401408 * 32 * 4);
  float* bufA   = (float*)carve((size_t)401408 * 64 * 4);
  float* pooled = (float*)carve((size_t)32 * 1024 * 4);
  if (cur > ws_size) return;

  k_init_slots<<<1, 64, 0, stream>>>(slots);
  k_bnfold<<<1, 64, 0, stream>>>((const float*)d_in[2], (const float*)d_in[3],
                                 (const float*)d_in[4], (const float*)d_in[5],
                                 headInv, headBias, 32);

  for (int j = 0; j < 13; ++j) {
    const float* dw_w = (const float*)d_in[6 + j * 10 + 0];
    const float* dg   = (const float*)d_in[6 + j * 10 + 1];
    const float* db   = (const float*)d_in[6 + j * 10 + 2];
    const float* dm   = (const float*)d_in[6 + j * 10 + 3];
    const float* dv   = (const float*)d_in[6 + j * 10 + 4];
    const float* pw_w = (const float*)d_in[6 + j * 10 + 5];
    const float* pg   = (const float*)d_in[6 + j * 10 + 6];
    const float* pb   = (const float*)d_in[6 + j * 10 + 7];
    const float* pm   = (const float*)d_in[6 + j * 10 + 8];
    const float* pv   = (const float*)d_in[6 + j * 10 + 9];
    int cin = CIN[j], cout = COUT[j];
    k_wq_dw<<<(cin + 63) / 64, 64, 0, stream>>>(dw_w, dg, db, dm, dv,
        dwCode[j], dwScale[j], dwBias[j], cin, OB_IN[j]);
    k_wq_pw<<<(cout + 63) / 64, 64, 0, stream>>>(pw_w, pg, pb, pm, pv,
        pwCode[j], pwScale[j], pwBias[j], cout, cin, KPAD[j], OB_OUT[j]);
  }

  {
    int total = NB * 112 * 112 * 32;
    k_head<<<(total + 255) / 256, 256, 0, stream>>>(x, head_w, headInv, headBias,
                                                    bufA, &slots[0], total);
  }

  int S = 112;
  for (int j = 0; j < 13; ++j) {
    int cin = CIN[j], cout = COUT[j], st = STR[j], kp = KPAD[j];
    int IH = S, IW = S, OH = S / st, OW = S / st;
    long long Pin  = (long long)NB * IH * IW;
    long long Pout = (long long)NB * OH * OW;

    {
      long long total = Pin * kp;
      k_quant<<<(int)((total + 255) / 256), 256, 0, stream>>>(
          bufA, &slots[2 * j], bufQ, cin, kp, total);
    }
    {
      long long total = Pout * cin;
      k_dw<<<(int)((total + 255) / 256), 256, 0, stream>>>(
          bufQ, dwCode[j], dwScale[j], dwBias[j], &slots[2 * j],
          bufB, &slots[2 * j + 1], cin, kp, IH, IW, OH, OW, st, total);
    }
    {
      long long total = Pout * kp;
      k_quant<<<(int)((total + 255) / 256), 256, 0, stream>>>(
          bufB, &slots[2 * j + 1], bufQ, cin, kp, total);
    }
    {
      int nTiles = (int)(Pout / 16);
      dim3 grid(cout / 16, (nTiles + 3) / 4);
      dim3 block(32, 4);
      k_pw_wmma<<<grid, block, 0, stream>>>(bufQ, pwCode[j], pwScale[j], pwBias[j],
          &slots[2 * j + 1], bufA, &slots[2 * j + 2], cout, kp, nTiles);
    }
    S = OH;
  }

  k_pool<<<(32 * 1024 + 255) / 256, 256, 0, stream>>>(bufA, pooled);
  k_fc<<<(32000 + 255) / 256, 256, 0, stream>>>(pooled, (const float*)d_in[136],
                                                (const float*)d_in[137], (float*)d_out);
}
